// E_MHSA_Original_20255065768207
// MI455X (gfx1250) — matrix operand order repaired and checked by reading, not compiled, not run on hardware
//
#include <hip/hip_runtime.h>

#define Bb    32
#define Cc    384
#define Nn    1024
#define NKk   256
#define NHEAD 12
#define HD    32
#define EPSF  1e-5f

typedef __bf16 bf16_t;
typedef __attribute__((ext_vector_type(16))) __bf16 v16bf;
typedef __attribute__((ext_vector_type(8)))  __bf16 v8bf;
typedef __attribute__((ext_vector_type(8)))  float  v8f;
typedef __attribute__((ext_vector_type(4)))  float  v4f;
typedef unsigned int __attribute__((ext_vector_type(4))) u32x4;
typedef int  __attribute__((ext_vector_type(4))) i32x4;
typedef int  __attribute__((ext_vector_type(8))) i32x8;

typedef __attribute__((address_space(1))) i32x4 gv4i_t;  // global <4 x i32>
typedef __attribute__((address_space(3))) i32x4 lv4i_t;  // LDS    <4 x i32>
typedef __attribute__((address_space(3))) void  lvoid_t;

// ---------------- CDNA5 async / tensor data-mover paths (guarded) ----------------
#if __has_builtin(__builtin_amdgcn_global_load_async_to_lds_b128)
#define HAVE_ASYNC_LDS 1
__device__ __forceinline__ void async_copy16(const void* gsrc, void* ldst) {
  __builtin_amdgcn_global_load_async_to_lds_b128((gv4i_t*)gsrc, (lv4i_t*)ldst, 0, 0);
}
__device__ __forceinline__ void async_wait0() {
#if __has_builtin(__builtin_amdgcn_s_wait_asynccnt)
  __builtin_amdgcn_s_wait_asynccnt(0);
#else
  asm volatile("s_wait_asynccnt 0x0" ::: "memory");
#endif
}
#else
#define HAVE_ASYNC_LDS 0
#endif

#if __has_builtin(__builtin_amdgcn_tensor_load_to_lds)
#define HAVE_TDM 1
#if __has_include(<hip/amd_detail/amd_gfx1250_TDM.h>)
#define TDM_SIX_ARG 1
#else
#define TDM_SIX_ARG 0
#endif
__device__ __forceinline__ void tdm_load_2d(u32x4 g0, i32x8 g1) {
  i32x4 z4 = {0, 0, 0, 0};
#if TDM_SIX_ARG
  i32x8 z8 = {0, 0, 0, 0, 0, 0, 0, 0};
  __builtin_amdgcn_tensor_load_to_lds(g0, g1, z4, z4, z8, 0);
#else
  __builtin_amdgcn_tensor_load_to_lds(g0, g1, z4, z4, 0);
#endif
}
__device__ __forceinline__ void tensor_wait0() {
#if __has_builtin(__builtin_amdgcn_s_wait_tensorcnt)
  __builtin_amdgcn_s_wait_tensorcnt(0);
#else
  asm volatile("s_wait_tensorcnt 0x0" ::: "memory");
#endif
}
#else
#define HAVE_TDM 0
#endif

__device__ __forceinline__ unsigned int lds_addr_of(const void* p) {
  return (unsigned int)(size_t)(lvoid_t*)p;
}

__device__ __forceinline__ v8f zero_v8f() {
  v8f z = {0.f, 0.f, 0.f, 0.f, 0.f, 0.f, 0.f, 0.f};
  return z;
}

// A-tile (16xK=32, bf16): lane l -> row (l&15), K elements {kb..kb+7, kb+16..kb+23}, kb=(l>>4)*8
__device__ __forceinline__ v16bf load_tile_a(const bf16_t* base, int ld) {
  const int lane = threadIdx.x & 31;
  const bf16_t* p = base + (lane & 15) * ld + ((lane >> 4) << 3);
  v8bf lo = *(const v8bf*)p;
  v8bf hi = *(const v8bf*)(p + 16);
  v16bf r;
#pragma unroll
  for (int i = 0; i < 8; ++i) { r[i] = lo[i]; r[i + 8] = hi[i]; }
  return r;
}

// B-tile (K=32 x 16, bf16) from a [16, >=32] row-major panel where row = output column:
// lane l -> column (l&15), K elements {kb..kb+7, kb+16..kb+23}, kb=(l>>4)*8
__device__ __forceinline__ v16bf load_tile_b(const bf16_t* base, int ld) {
  const int lane = threadIdx.x & 31;
  const bf16_t* p = base + (lane & 15) * ld + ((lane >> 4) << 3);
  v8bf lo = *(const v8bf*)p;
  v8bf hi = *(const v8bf*)(p + 16);
  v16bf r;
#pragma unroll
  for (int i = 0; i < 8; ++i) { r[i] = lo[i]; r[i + 8] = hi[i]; }
  return r;
}

__device__ __forceinline__ v8f wmma_bf16(v16bf a, v16bf b, v8f c) {
  return __builtin_amdgcn_wmma_f32_16x16x32_bf16(false, a, false, b, (short)0, c,
                                                 false, false);
}

// ---- prep: W[Cin,Cout] f32 -> Wt[Cout,Cin] bf16 ----
__global__ __launch_bounds__(256) void wt_kernel(const float* __restrict__ W,
                                                 bf16_t* __restrict__ Wt) {
  int t = blockIdx.x * 256 + threadIdx.x;
  if (t >= Cc * Cc) return;
  int co = t / Cc, ci = t % Cc;
  Wt[co * Cc + ci] = (bf16_t)W[ci * Cc + co];
}

// ---- prep: x[B,C,N] f32 -> tok[B*N, C] bf16 ----
__global__ __launch_bounds__(256) void tok_kernel(const float* __restrict__ x,
                                                  bf16_t* __restrict__ tok) {
  int t = blockIdx.x * 256 + threadIdx.x;
  if (t >= Bb * Cc * Nn) return;
  int n = t % Nn;
  int c = (t / Nn) % Cc;
  int b = t / (Nn * Cc);
  tok[((size_t)(b * Nn + n)) * Cc + c] = (bf16_t)x[t];
}

// ---- prep: avgpool(4)+BN -> xr[B*Nk, C] bf16 ----
__global__ __launch_bounds__(256) void pool_bn_kernel(
    const float* __restrict__ x, const float* __restrict__ gamma,
    const float* __restrict__ beta, const float* __restrict__ mean,
    const float* __restrict__ var, bf16_t* __restrict__ xr) {
  int t = blockIdx.x * 256 + threadIdx.x;
  if (t >= Bb * Cc * NKk) return;
  int m = t % NKk;
  int c = (t / NKk) % Cc;
  int b = t / (NKk * Cc);
  const float4 x4 = *(const float4*)(x + ((size_t)(b * Cc + c)) * Nn + 4 * m);
  float xp = 0.25f * (x4.x + x4.y + x4.z + x4.w);
  float inv = gamma[c] * rsqrtf(var[c] + EPSF);
  float val = xp * inv + (beta[c] - mean[c] * inv);
  xr[((size_t)(b * NKk + m)) * Cc + c] = (bf16_t)val;
}

// ---- stage a 64x384 bf16 weight panel into LDS (async DMA when available) ----
__device__ __forceinline__ void stage_w_panel(const bf16_t* __restrict__ Wt,
                                              int colBase, bf16_t (*w_s)[Cc]) {
  const bf16_t* wsrc = Wt + (size_t)colBase * Cc;
  for (int i = threadIdx.x; i < (64 * Cc) / 8; i += 256) {
    int r = i / (Cc / 8);
    int c8 = (i % (Cc / 8)) * 8;
    const bf16_t* g = wsrc + (size_t)r * Cc + c8;
    bf16_t* l = &w_s[r][c8];
#if HAVE_ASYNC_LDS
    async_copy16(g, l);
#else
    *(v8bf*)l = *(const v8bf*)g;
#endif
  }
#if HAVE_ASYNC_LDS
  async_wait0();
#endif
  __syncthreads();
}

// ---- generic GEMM: Out[M,384] = A[M,384] @ Wt^T + bias, bf16 in/out, f32 acc ----
__global__ __launch_bounds__(256) void gemm_kernel(const bf16_t* __restrict__ A,
                                                   const bf16_t* __restrict__ Wt,
                                                   const float* __restrict__ bias,
                                                   bf16_t* __restrict__ Out) {
  __shared__ __align__(32) bf16_t w_s[64][Cc];  // 48 KB weight panel
  const int wid = threadIdx.x >> 5;
  const int lane = threadIdx.x & 31;
  const int l15 = lane & 15, lhi = lane >> 4;
  const int rowBase = blockIdx.x * 128 + wid * 16;
  const int colBase = blockIdx.y * 64;

  stage_w_panel(Wt, colBase, w_s);

  const bf16_t* Arow = A + (size_t)rowBase * Cc;
  v8f acc[4];
#pragma unroll
  for (int j = 0; j < 4; ++j) acc[j] = zero_v8f();
  for (int k0 = 0; k0 < Cc; k0 += 32) {
    if (k0 + 32 < Cc) __builtin_prefetch(Arow + k0 + 32, 0, 3);
    v16bf a = load_tile_a(Arow + k0, Cc);
#pragma unroll
    for (int j = 0; j < 4; ++j) {
      v16bf bm = load_tile_b(&w_s[j * 16][k0], Cc);
      acc[j] = wmma_bf16(a, bm, acc[j]);
    }
  }
#pragma unroll
  for (int j = 0; j < 4; ++j) {
    int col = colBase + j * 16 + l15;
    float bval = bias[col];
    int r0 = rowBase + (lhi << 3);
#pragma unroll
    for (int r = 0; r < 8; ++r)
      Out[(size_t)(r0 + r) * Cc + col] = (bf16_t)(acc[j][r] + bval);
  }
}

// ---- attention: per (b, head, 64-row strip); 4 waves, one 16-row tile each ----
__global__ __launch_bounds__(128) void attn_kernel(const bf16_t* __restrict__ Q,
                                                   const bf16_t* __restrict__ Kb,
                                                   const bf16_t* __restrict__ Vb,
                                                   bf16_t* __restrict__ O) {
  __shared__ __align__(32) bf16_t k_s[NKk][HD];     // 16 KB
  __shared__ __align__(32) bf16_t vT_s[HD][NKk];    // 16 KB
  __shared__ __align__(32) bf16_t p_s[4][16][NKk];  // 32 KB

  const int id = blockIdx.x;
  const int tile = id & 15;
  const int head = (id >> 4) % NHEAD;
  const int b = id / (16 * NHEAD);
  const int tid = threadIdx.x;

  // ---- stage K tile [256 x 32] bf16 (row stride 384) ----
#if HAVE_TDM
  if (tid < 32) {  // wave 0 issues one 2-D TDM descriptor, then waits TENSORcnt
    unsigned long long ga =
        (unsigned long long)(size_t)(Kb + ((size_t)b * NKk) * Cc + head * HD);
    u32x4 g0;
    g0[0] = 1u;                                        // count=1, user mode
    g0[1] = lds_addr_of(&k_s[0][0]);                   // lds_addr
    g0[2] = (unsigned int)ga;                          // global_addr[31:0]
    g0[3] = (unsigned int)((ga >> 32) & 0x01FFFFFFu)   // global_addr[56:32]
            | (2u << 30);                              // type=2 (image)
    i32x8 g1;
    g1[0] = (int)(1u << 16);    // data_size=1 -> 2-byte elements
    g1[1] = (int)(32u << 16);   // tensor_dim0 = 32
    g1[2] = (int)(256u << 16);  // tensor_dim1 = 256
    g1[3] = (int)(32u << 16);   // tile_dim0 = 32
    g1[4] = (int)256;           // tile_dim1 = 256
    g1[5] = (int)384;           // tensor_dim0_stride = 384 elements
    g1[6] = 0;
    g1[7] = 0;
    tdm_load_2d(g0, g1);
    tensor_wait0();
  }
#else
  for (int i = tid; i < (NKk * HD) / 8; i += 128) {
    int m = i >> 2, dc = (i & 3) << 3;
    const bf16_t* g = Kb + ((size_t)(b * NKk + m)) * Cc + head * HD + dc;
#if HAVE_ASYNC_LDS
    async_copy16(g, &k_s[m][dc]);
#else
    *(v8bf*)&k_s[m][dc] = *(const v8bf*)g;
#endif
  }
#if HAVE_ASYNC_LDS
  async_wait0();
#endif
#endif

  // ---- stage V transposed [32 x 256] bf16 (transpose in the move) ----
  for (int i = tid; i < (NKk * HD) / 8; i += 128) {
    int m = i >> 2, dc = (i & 3) << 3;
    v8bf v = *(const v8bf*)(Vb + ((size_t)(b * NKk + m)) * Cc + head * HD + dc);
#pragma unroll
    for (int e = 0; e < 8; ++e) vT_s[dc + e][m] = v[e];
  }
  __syncthreads();

  const int wid = tid >> 5;
  const int lane = tid & 31;
  const int l15 = lane & 15, lhi = lane >> 4;
  const int n0 = tile * 64 + wid * 16;

  // scores strip: 16 rows x 256 cols, K = d = 32 -> one WMMA per 16x16 tile
  v16bf aq = load_tile_a(Q + ((size_t)(b * Nn + n0)) * Cc + head * HD, Cc);
  v8f s[16];
#pragma unroll
  for (int t = 0; t < 16; ++t) {
    v16bf bk = load_tile_b(&k_s[t * 16][0], HD);
    s[t] = wmma_bf16(aq, bk, zero_v8f());
  }

  // softmax over 256 (scale = 1/sqrt(32)); row lives in one VGPR idx across 16 lanes
  const float scale = 0.17677669529663687f;
#pragma unroll
  for (int r = 0; r < 8; ++r) {
    float mx = -3.0e38f;
#pragma unroll
    for (int t = 0; t < 16; ++t) mx = fmaxf(mx, s[t][r]);
    mx = fmaxf(mx, __shfl_xor(mx, 1, 16));
    mx = fmaxf(mx, __shfl_xor(mx, 2, 16));
    mx = fmaxf(mx, __shfl_xor(mx, 4, 16));
    mx = fmaxf(mx, __shfl_xor(mx, 8, 16));
    float sum = 0.f;
#pragma unroll
    for (int t = 0; t < 16; ++t) {
      float e = __expf((s[t][r] - mx) * scale);
      s[t][r] = e;
      sum += e;
    }
    sum += __shfl_xor(sum, 1, 16);
    sum += __shfl_xor(sum, 2, 16);
    sum += __shfl_xor(sum, 4, 16);
    sum += __shfl_xor(sum, 8, 16);
    float rinv = 1.0f / sum;
    int row = (lhi << 3) + r;
#pragma unroll
    for (int t = 0; t < 16; ++t)
      p_s[wid][row][t * 16 + l15] = (bf16_t)(s[t][r] * rinv);
  }
  // same-wave LDS RAW: DS ops are in-order per wave

  // out = P[16x256] @ V[256x32] : 8 K-chunks x 2 N-tiles
  v8f o0 = zero_v8f(), o1 = zero_v8f();
#pragma unroll
  for (int kt = 0; kt < 8; ++kt) {
    v16bf ap = load_tile_a(&p_s[wid][0][kt * 32], NKk);
    v16bf b0 = load_tile_b(&vT_s[0][kt * 32], NKk);
    v16bf b1 = load_tile_b(&vT_s[16][kt * 32], NKk);
    o0 = wmma_bf16(ap, b0, o0);
    o1 = wmma_bf16(ap, b1, o1);
  }
  {
    int rbase = b * Nn + n0 + (lhi << 3);
#pragma unroll
    for (int r = 0; r < 8; ++r) {
      size_t rowoff = (size_t)(rbase + r) * Cc + head * HD;
      O[rowoff + l15] = (bf16_t)o0[r];
      O[rowoff + 16 + l15] = (bf16_t)o1[r];
    }
  }
}

// ---- proj GEMM, writes fp32 transposed into [B, C, H*W] ----
__global__ __launch_bounds__(256) void proj_kernel(const bf16_t* __restrict__ A,
                                                   const bf16_t* __restrict__ Wt,
                                                   const float* __restrict__ bias,
                                                   float* __restrict__ OutF) {
  __shared__ __align__(32) bf16_t w_s[64][Cc];
  const int wid = threadIdx.x >> 5;
  const int lane = threadIdx.x & 31;
  const int l15 = lane & 15, lhi = lane >> 4;
  const int rowBase = blockIdx.x * 128 + wid * 16;
  const int colBase = blockIdx.y * 64;

  stage_w_panel(Wt, colBase, w_s);

  const bf16_t* Arow = A + (size_t)rowBase * Cc;
  v8f acc[4];
#pragma unroll
  for (int j = 0; j < 4; ++j) acc[j] = zero_v8f();
  for (int k0 = 0; k0 < Cc; k0 += 32) {
    if (k0 + 32 < Cc) __builtin_prefetch(Arow + k0 + 32, 0, 3);
    v16bf a = load_tile_a(Arow + k0, Cc);
#pragma unroll
    for (int j = 0; j < 4; ++j) {
      v16bf bm = load_tile_b(&w_s[j * 16][k0], Cc);
      acc[j] = wmma_bf16(a, bm, acc[j]);
    }
  }
#pragma unroll
  for (int j = 0; j < 4; ++j) {
    int col = colBase + j * 16 + l15;
    float bval = bias[col];
    int grow = rowBase + (lhi << 3);
    int bb = grow >> 10;    // batch
    int nn0 = grow & 1023;  // token index, consecutive across acc rows
    float* dst = OutF + (((size_t)(bb * Cc + col)) << 10) + nn0;
    v4f lo = {acc[j][0] + bval, acc[j][1] + bval, acc[j][2] + bval, acc[j][3] + bval};
    v4f hi = {acc[j][4] + bval, acc[j][5] + bval, acc[j][6] + bval, acc[j][7] + bval};
    *(v4f*)dst = lo;
    *(v4f*)(dst + 4) = hi;
  }
}

extern "C" void kernel_launch(void* const* d_in, const int* in_sizes, int n_in,
                              void* d_out, int out_size, void* d_ws, size_t ws_size,
                              hipStream_t stream) {
  const float* x     = (const float*)d_in[0];
  const float* Wq    = (const float*)d_in[1];
  const float* bq    = (const float*)d_in[2];
  const float* Wk    = (const float*)d_in[3];
  const float* bk    = (const float*)d_in[4];
  const float* Wv    = (const float*)d_in[5];
  const float* bv    = (const float*)d_in[6];
  const float* Wp    = (const float*)d_in[7];
  const float* bp    = (const float*)d_in[8];
  const float* gamma = (const float*)d_in[9];
  const float* beta  = (const float*)d_in[10];
  const float* mean  = (const float*)d_in[11];
  const float* var   = (const float*)d_in[12];
  float* out = (float*)d_out;
  (void)in_sizes; (void)n_in; (void)out_size; (void)ws_size;

  char* ws = (char*)d_ws;
  size_t off = 0;
  auto carve = [&](size_t bytes) -> char* {
    char* p = ws + off;
    off += (bytes + 255) & ~(size_t)255;
    return p;
  };
  bf16_t* tok = (bf16_t*)carve((size_t)Bb * Nn * Cc * 2);   // reused as attn output
  bf16_t* Qb  = (bf16_t*)carve((size_t)Bb * Nn * Cc * 2);
  bf16_t* xr  = (bf16_t*)carve((size_t)Bb * NKk * Cc * 2);
  bf16_t* Kb  = (bf16_t*)carve((size_t)Bb * NKk * Cc * 2);
  bf16_t* Vb  = (bf16_t*)carve((size_t)Bb * NKk * Cc * 2);
  bf16_t* Wtq = (bf16_t*)carve((size_t)Cc * Cc * 2);
  bf16_t* Wtk = (bf16_t*)carve((size_t)Cc * Cc * 2);
  bf16_t* Wtv = (bf16_t*)carve((size_t)Cc * Cc * 2);
  bf16_t* Wtp = (bf16_t*)carve((size_t)Cc * Cc * 2);

  dim3 blk(256);
  int wtBlocks = (Cc * Cc + 255) / 256;
  wt_kernel<<<wtBlocks, blk, 0, stream>>>(Wq, Wtq);
  wt_kernel<<<wtBlocks, blk, 0, stream>>>(Wk, Wtk);
  wt_kernel<<<wtBlocks, blk, 0, stream>>>(Wv, Wtv);
  wt_kernel<<<wtBlocks, blk, 0, stream>>>(Wp, Wtp);

  tok_kernel<<<(Bb * Cc * Nn + 255) / 256, blk, 0, stream>>>(x, tok);
  pool_bn_kernel<<<(Bb * Cc * NKk + 255) / 256, blk, 0, stream>>>(x, gamma, beta,
                                                                  mean, var, xr);

  gemm_kernel<<<dim3(Bb * Nn / 128, Cc / 64), blk, 0, stream>>>(tok, Wtq, bq, Qb);
  gemm_kernel<<<dim3(Bb * NKk / 128, Cc / 64), blk, 0, stream>>>(xr, Wtk, bk, Kb);
  gemm_kernel<<<dim3(Bb * NKk / 128, Cc / 64), blk, 0, stream>>>(xr, Wtv, bv, Vb);

  bf16_t* Oattn = tok;  // tok no longer needed after Q projection
  attn_kernel<<<dim3(Bb * NHEAD * 16), dim3(128), 0, stream>>>(Qb, Kb, Vb, Oattn);

  proj_kernel<<<dim3(Bb * Nn / 128, Cc / 64), blk, 0, stream>>>(Oattn, Wtp, bp, out);
}
